// TransformerBlock_61598420959848
// MI455X (gfx1250) — matrix operand order repaired and checked by reading, not compiled, not run on hardware
//
#include <hip/hip_runtime.h>
#include <math.h>

// ---------------------------------------------------------------------------
// Types
// ---------------------------------------------------------------------------
typedef unsigned short u16;
typedef __attribute__((ext_vector_type(16))) __bf16        bf16x16;
typedef __attribute__((ext_vector_type(8)))  float         f32x8;
typedef __attribute__((ext_vector_type(4)))  unsigned int  u32x4;
typedef __attribute__((ext_vector_type(2)))  unsigned int  u32x2;
typedef int v4i_vs __attribute__((vector_size(16)));   // matches builtin's V4i param

// CDNA5 async global->LDS path (ASYNCcnt-tracked), with sync fallback.
#if defined(__has_builtin)
#if __has_builtin(__builtin_amdgcn_global_load_async_to_lds_b128) && \
    __has_builtin(__builtin_amdgcn_s_wait_asynccnt)
#define USE_ASYNC_LDS 1
#endif
#endif

static __device__ __forceinline__ void async_or_sync_b128(u16* lds_dst, const u16* gsrc) {
#if defined(USE_ASYNC_LDS)
    __builtin_amdgcn_global_load_async_to_lds_b128(
        (__attribute__((address_space(1))) v4i_vs*)gsrc,
        (__attribute__((address_space(3))) v4i_vs*)lds_dst,
        /*offset=*/0, /*cpol=*/0);
#else
    *(u32x4*)lds_dst = *(const u32x4*)gsrc;
#endif
}

static __device__ __forceinline__ void wait_async_lds() {
#if defined(USE_ASYNC_LDS)
    __builtin_amdgcn_s_wait_asynccnt(0);
#endif
}

// fp32 -> bf16 round-to-nearest-even, stored as raw u16
static __device__ __forceinline__ u16 f2bf(float f) {
    unsigned u = __float_as_uint(f);
    unsigned r = u + 0x7FFFu + ((u >> 16) & 1u);
    return (u16)(r >> 16);
}

// Load a 32-byte WMMA fragment as two 16B chunks (ds/global b128)
static __device__ __forceinline__ bf16x16 ld_frag(const u16* p0, const u16* p1) {
    union U { bf16x16 v; u32x4 q[2]; } u;
    u.q[0] = *(const u32x4*)p0;
    u.q[1] = *(const u32x4*)p1;
    return u.v;
}

static __device__ __forceinline__ f32x8 wmma_bf16(bf16x16 a, bf16x16 b, f32x8 c) {
    return __builtin_amdgcn_wmma_f32_16x16x32_bf16(
        /*neg_a=*/false, a, /*neg_b=*/false, b,
        /*c_mod=*/(short)0, c, /*reuse_a=*/false, /*reuse_b=*/false);
}

static __device__ __forceinline__ f32x8 fzero8() {
    f32x8 z = {0.f,0.f,0.f,0.f,0.f,0.f,0.f,0.f};
    return z;
}

// ---------------------------------------------------------------------------
// Model constants
// ---------------------------------------------------------------------------
#define C_DIM   768
#define C3_DIM  2304
#define F_DIM   3072
#define H_NUM   12
#define DH      64
#define T_LEN   2048
#define B_NUM   2
#define M_TOK   (B_NUM * T_LEN)   // 4096

// ---------------------------------------------------------------------------
// fp32 -> bf16 cast kernel (for weights)
// ---------------------------------------------------------------------------
__global__ void cast_bf16_kernel(const float* __restrict__ in, u16* __restrict__ out, int n) {
    int i = blockIdx.x * blockDim.x + threadIdx.x;
    if (i < n) out[i] = f2bf(in[i]);
}

// ---------------------------------------------------------------------------
// LayerNorm over rows of length 768 -> bf16 output. One block (256 thr) per row.
// ---------------------------------------------------------------------------
__global__ __launch_bounds__(256) void ln_kernel(const float* __restrict__ x,
                                                 const float* __restrict__ g,
                                                 const float* __restrict__ b,
                                                 u16* __restrict__ out) {
    int row = blockIdx.x;
    const float* xr = x + (size_t)row * C_DIM;
    float v[3];
    float s = 0.f, s2 = 0.f;
#pragma unroll
    for (int i = 0; i < 3; ++i) {
        v[i] = xr[threadIdx.x + i * 256];
        s  += v[i];
        s2 += v[i] * v[i];
    }
    __shared__ float rs[256], rq[256];
    rs[threadIdx.x] = s; rq[threadIdx.x] = s2;
    __syncthreads();
    for (int st = 128; st > 0; st >>= 1) {
        if ((int)threadIdx.x < st) {
            rs[threadIdx.x] += rs[threadIdx.x + st];
            rq[threadIdx.x] += rq[threadIdx.x + st];
        }
        __syncthreads();
    }
    float mu  = rs[0] * (1.0f / C_DIM);
    float var = rq[0] * (1.0f / C_DIM) - mu * mu;
    float inv = rsqrtf(var + 1e-5f);
#pragma unroll
    for (int i = 0; i < 3; ++i) {
        int c = threadIdx.x + i * 256;
        out[(size_t)row * C_DIM + c] = f2bf((v[i] - mu) * inv * g[c] + b[c]);
    }
}

// ---------------------------------------------------------------------------
// bf16 GEMM: C[M,N] = A[M,K] * B[K,N], fp32 accumulate via v_wmma_f32_16x16x32_bf16
// Block tile 128x128x32, 256 threads = 8 waves; wave tile 64(M) x 32(N).
// Double-buffered LDS: A tile staged with global_load_async_to_lds_b128,
// B tile prefetched to VGPRs and transpose-scattered to LDS; one barrier/K-step.
// Epilogue is compile-time configured (no uniform runtime branching).
// ---------------------------------------------------------------------------
#define GBM 128
#define GBN 128
#define GBK 32
#define LDT 40   // padded LDS stride (halfwords): 80B rows keep 16B alignment

template<bool HAS_BIAS, bool DO_GELU, bool HAS_RES, bool OUT_F, bool OUT_B>
__global__ __launch_bounds__(256) void gemm_bf16_kernel(
        const u16* __restrict__ A, const u16* __restrict__ Bm,
        int N, int K,
        const float* __restrict__ bias,
        const float* __restrict__ res,
        float* __restrict__ outF, u16* __restrict__ outB) {
    __shared__ __align__(16) u16 lA[2][GBM * LDT];
    __shared__ __align__(16) u16 lB[2][GBN * LDT];   // transposed: [n][k]

    const int tid  = threadIdx.x;
    const int lane = tid & 31;
    const int wv   = tid >> 5;
    const int wm   = wv & 1;    // 0..1  -> 64 rows each
    const int wn   = wv >> 1;   // 0..3  -> 32 cols each
    const int rl   = lane & 15;
    const int hi   = lane >> 4;
    const int kb8  = hi * 8;    // A frag K base
    const int kh   = hi * 16;   // unused; B frags also use kb8

    const int bm0 = blockIdx.y * GBM;
    const int bn0 = blockIdx.x * GBN;

    // A-tile staging: 512 chunks of 8 bf16 (async b128 direct to LDS)
    auto stageA = [&](int kb, int p) {
#pragma unroll
        for (int i = 0; i < 2; ++i) {
            int c   = tid * 2 + i;
            int r   = c >> 2;
            int off = (c & 3) * 8;
            async_or_sync_b128(&lA[p][r * LDT + off],
                               A + (size_t)(bm0 + r) * K + kb + off);
        }
    };
    // B-tile global -> regs (1024 chunks of 4 bf16)
    auto loadB = [&](int kb, u32x2* breg) {
#pragma unroll
        for (int i = 0; i < 4; ++i) {
            int c  = tid * 4 + i;
            int kr = c >> 5;
            int nc = (c & 31) * 4;
            breg[i] = *(const u32x2*)(Bm + (size_t)(kb + kr) * N + bn0 + nc);
        }
    };
    // B-tile regs -> LDS transposed
    auto storeB = [&](int p, const u32x2* breg) {
#pragma unroll
        for (int i = 0; i < 4; ++i) {
            int c  = tid * 4 + i;
            int kr = c >> 5;
            int nc = (c & 31) * 4;
            union { u32x2 u; u16 h[4]; } t;
            t.u = breg[i];
#pragma unroll
            for (int j = 0; j < 4; ++j) lB[p][(nc + j) * LDT + kr] = t.h[j];
        }
    };

    f32x8 acc[4][2];
#pragma unroll
    for (int mi = 0; mi < 4; ++mi)
#pragma unroll
        for (int ni = 0; ni < 2; ++ni) acc[mi][ni] = fzero8();

    // ---- pipeline prologue: stage tile 0 into buffer 0 ----
    u32x2 breg[4];
    stageA(0, 0);
    loadB(0, breg);
    storeB(0, breg);
    wait_async_lds();
    __syncthreads();

    int p = 0;
    for (int kb = 0; kb < K; kb += GBK, p ^= 1) {
        const bool more = (kb + GBK) < K;
        if (more) {
            stageA(kb + GBK, p ^ 1);   // async into other buffer
            loadB(kb + GBK, breg);     // global -> regs (overlaps compute)
        }

        // ---- fragments from buffer p ----
        bf16x16 af[4], bfrag[2];
#pragma unroll
        for (int mi = 0; mi < 4; ++mi) {
            const u16* q = &lA[p][(wm * 64 + mi * 16 + rl) * LDT + kb8];
            af[mi] = ld_frag(q, q + 16);
        }
#pragma unroll
        for (int ni = 0; ni < 2; ++ni) {
            const u16* q = &lB[p][(wn * 32 + ni * 16 + rl) * LDT + kb8];
            bfrag[ni] = ld_frag(q, q + 16);
        }
#pragma unroll
        for (int mi = 0; mi < 4; ++mi)
#pragma unroll
            for (int ni = 0; ni < 2; ++ni)
                acc[mi][ni] = wmma_bf16(af[mi], bfrag[ni], acc[mi][ni]);

        if (more) {
            storeB(p ^ 1, breg);       // regs -> LDS (other buffer)
            wait_async_lds();          // A-tile asyncs landed before barrier
        }
        __syncthreads();
    }

    // ---- straight-line epilogue (compile-time configured) ----
#pragma unroll
    for (int mi = 0; mi < 4; ++mi) {
#pragma unroll
        for (int ni = 0; ni < 2; ++ni) {
#pragma unroll
            for (int r = 0; r < 8; ++r) {
                int row = bm0 + wm * 64 + mi * 16 + r + hi * 8;
                int col = bn0 + wn * 32 + ni * 16 + rl;
                float v = acc[mi][ni][r];
                if constexpr (HAS_BIAS) v += bias[col];
                if constexpr (DO_GELU)  v = 0.5f * v * (1.0f + erff(v * 0.70710678118654752f));
                if constexpr (HAS_RES)  v += res[(size_t)row * N + col];
                if constexpr (OUT_F)    outF[(size_t)row * N + col] = v;
                if constexpr (OUT_B)    outB[(size_t)row * N + col] = f2bf(v);
            }
        }
    }
}

// ---------------------------------------------------------------------------
// Causal flash attention: grid (B*H, T/64), 128 threads = 4 waves.
// Each wave owns 16 query rows; key blocks of 32 staged in LDS and shared.
// K tile staged via async global->LDS; V tile transposed through VGPRs.
// qkv: bf16 [M_TOK][2304] with Q|K|V column blocks; out: bf16 [M_TOK][768].
// ---------------------------------------------------------------------------
__global__ __launch_bounds__(128) void attn_kernel(const u16* __restrict__ qkv,
                                                   u16* __restrict__ out) {
    const int bh = blockIdx.x;
    const int b  = bh / H_NUM;
    const int h  = bh % H_NUM;
    const int qb = blockIdx.y;          // 64-row query block

    const int tid  = threadIdx.x;
    const int lane = tid & 31;
    const int wv   = tid >> 5;
    const int rl   = lane & 15;
    const int hi   = lane >> 4;
    const int kb8  = hi * 8;
    const int kh   = hi * 16;

    const size_t tok0 = (size_t)b * T_LEN;

    __shared__ __align__(16) u16 sK[32 * 64];      // [key][d]
    __shared__ __align__(16) u16 sV[64 * 32];      // [d][key] (transposed)
    __shared__ __align__(16) u16 sP[4][16 * 32];   // per-wave P tile

    // ---- Q fragments (A layout), loaded once ----
    const int qrow_lane = qb * 64 + wv * 16 + rl;
    const u16* qptr = qkv + (tok0 + qrow_lane) * C3_DIM + h * DH;
    bf16x16 qf[2];
    qf[0] = ld_frag(qptr + kb8,      qptr + 16 + kb8);
    qf[1] = ld_frag(qptr + 32 + kb8, qptr + 48 + kb8);

    f32x8 accO[4];
#pragma unroll
    for (int n = 0; n < 4; ++n) accO[n] = fzero8();
    float mrow[8], lrow[8];
#pragma unroll
    for (int r = 0; r < 8; ++r) { mrow[r] = -3.0e38f; lrow[r] = 0.f; }

    const int nkb = qb * 2 + 2;   // key blocks of 32 covering causal range
    for (int kblk = 0; kblk < nkb; ++kblk) {
        __syncthreads();
        // ---- stage K block [32 keys][64 d] via async global->LDS ----
#pragma unroll
        for (int i = 0; i < 2; ++i) {
            int c   = tid + i * 128;     // 256 chunks of 8 bf16
            int kr  = c >> 3;
            int off = (c & 7) * 8;
            async_or_sync_b128(&sK[kr * 64 + off],
                qkv + (tok0 + kblk * 32 + kr) * C3_DIM + C_DIM + h * DH + off);
        }
        // ---- stage V block transposed -> sV[d][key] (overlaps K asyncs) ----
#pragma unroll
        for (int i = 0; i < 4; ++i) {
            int c  = tid + i * 128;      // 512 chunks of 4 bf16
            int kr = c >> 4;
            int dc = (c & 15) * 4;
            union { u32x2 u; u16 hh[4]; } t;
            t.u = *(const u32x2*)(qkv + (tok0 + kblk * 32 + kr) * C3_DIM + 2 * C_DIM + h * DH + dc);
#pragma unroll
            for (int j = 0; j < 4; ++j) sV[(dc + j) * 32 + kr] = t.hh[j];
        }
        wait_async_lds();
        __syncthreads();

        // ---- S = Q K^T for two 16-key subtiles ----
        f32x8 s[2];
#pragma unroll
        for (int kt = 0; kt < 2; ++kt) {
            f32x8 z = fzero8();
            const u16* kp0 = &sK[(kt * 16 + rl) * 64 + kb8];
            const u16* kp1 = &sK[(kt * 16 + rl) * 64 + 32 + kb8];
            z = wmma_bf16(qf[0], ld_frag(kp0, kp0 + 16), z);
            z = wmma_bf16(qf[1], ld_frag(kp1, kp1 + 16), z);
            s[kt] = z;
        }

        // ---- scale + causal mask + online softmax ----
        const int k0 = kblk * 32 + rl;
        float pmax[8];
#pragma unroll
        for (int r = 0; r < 8; ++r) {
            int qr = qb * 64 + wv * 16 + r + hi * 8;
            float v0 = s[0][r] * 0.125f; if (k0 > qr)      v0 = -3.0e38f;
            float v1 = s[1][r] * 0.125f; if (k0 + 16 > qr) v1 = -3.0e38f;
            s[0][r] = v0; s[1][r] = v1;
            pmax[r] = fmaxf(v0, v1);
        }
#pragma unroll
        for (int r = 0; r < 8; ++r) {
#pragma unroll
            for (int off = 8; off >= 1; off >>= 1)
                pmax[r] = fmaxf(pmax[r], __shfl_xor(pmax[r], off, 32));
        }
        float alpha[8], psum[8];
#pragma unroll
        for (int r = 0; r < 8; ++r) {
            float mn = fmaxf(mrow[r], pmax[r]);
            alpha[r] = __expf(mrow[r] - mn);
            mrow[r]  = mn;
            float p0 = __expf(s[0][r] - mn);
            float p1 = __expf(s[1][r] - mn);
            s[0][r] = p0; s[1][r] = p1;
            psum[r] = p0 + p1;
        }
#pragma unroll
        for (int r = 0; r < 8; ++r) {
#pragma unroll
            for (int off = 8; off >= 1; off >>= 1)
                psum[r] += __shfl_xor(psum[r], off, 32);
            lrow[r] = lrow[r] * alpha[r] + psum[r];
        }
#pragma unroll
        for (int n = 0; n < 4; ++n)
#pragma unroll
            for (int r = 0; r < 8; ++r) accO[n][r] *= alpha[r];

        // ---- P: C-layout fp32 -> A-layout bf16 via per-wave LDS ----
        u16* pw = &sP[wv][0];
#pragma unroll
        for (int r = 0; r < 8; ++r) {
            int prow = r + hi * 8;
            pw[prow * 32 + rl]      = f2bf(s[0][r]);
            pw[prow * 32 + 16 + rl] = f2bf(s[1][r]);
        }
        const u16* pr = &sP[wv][rl * 32 + kb8];
        bf16x16 pf = ld_frag(pr, pr + 16);   // same-wave DS ops are in-order

        // ---- O += P @ V ----
#pragma unroll
        for (int n = 0; n < 4; ++n) {
            const u16* vp = &sV[(n * 16 + rl) * 32 + kb8];
            accO[n] = wmma_bf16(pf, ld_frag(vp, vp + 16), accO[n]);
        }
    }

    // ---- normalize and store bf16 [token][h*64+d] ----
#pragma unroll
    for (int n = 0; n < 4; ++n) {
#pragma unroll
        for (int r = 0; r < 8; ++r) {
            int qr  = qb * 64 + wv * 16 + r + hi * 8;
            int col = h * DH + n * 16 + rl;
            out[(tok0 + qr) * C_DIM + col] = f2bf(accO[n][r] / lrow[r]);
        }
    }
}

// ---------------------------------------------------------------------------
// Host launcher
// ---------------------------------------------------------------------------
extern "C" void kernel_launch(void* const* d_in, const int* in_sizes, int n_in,
                              void* d_out, int out_size, void* d_ws, size_t ws_size,
                              hipStream_t stream) {
    (void)in_sizes; (void)n_in; (void)out_size; (void)ws_size;

    const float* x     = (const float*)d_in[0];
    const float* w_qkv = (const float*)d_in[1];
    const float* w_out = (const float*)d_in[2];
    const float* ln1_g = (const float*)d_in[3];
    const float* ln1_b = (const float*)d_in[4];
    const float* ln2_g = (const float*)d_in[5];
    const float* ln2_b = (const float*)d_in[6];
    const float* w1    = (const float*)d_in[7];
    const float* b1    = (const float*)d_in[8];
    const float* w2    = (const float*)d_in[9];
    const float* b2    = (const float*)d_in[10];
    float* out = (float*)d_out;

    char* ws = (char*)d_ws;
    size_t o = 0;
    // R1: xln / attn_out / ln2_out share one bf16 buffer (used sequentially)
    u16* actB  = (u16*)(ws + o); o += (size_t)M_TOK * C_DIM * 2;          // 6.29 MB
    // R2: qkv (first 18.9MB) / ffn hidden h1 share one region
    u16* qkvB  = (u16*)(ws + o);
    u16* h1B   = (u16*)(ws + o); o += (size_t)M_TOK * F_DIM * 2;          // 25.2 MB
    u16* wqkvB = (u16*)(ws + o); o += (size_t)C_DIM * C3_DIM * 2;         // 3.54 MB
    u16* woutB = (u16*)(ws + o); o += (size_t)C_DIM * C_DIM * 2;         // 1.18 MB
    u16* w1B   = (u16*)(ws + o); o += (size_t)C_DIM * F_DIM * 2;          // 4.72 MB
    u16* w2B   = (u16*)(ws + o); o += (size_t)F_DIM * C_DIM * 2;          // 4.72 MB
    float* x1  = (float*)(ws + o); o += (size_t)M_TOK * C_DIM * 4;        // 12.6 MB

    // 1) weights -> bf16
    {
        int n;
        n = C_DIM * C3_DIM; cast_bf16_kernel<<<(n + 255) / 256, 256, 0, stream>>>(w_qkv, wqkvB, n);
        n = C_DIM * C_DIM;  cast_bf16_kernel<<<(n + 255) / 256, 256, 0, stream>>>(w_out, woutB, n);
        n = C_DIM * F_DIM;  cast_bf16_kernel<<<(n + 255) / 256, 256, 0, stream>>>(w1,    w1B,   n);
        n = F_DIM * C_DIM;  cast_bf16_kernel<<<(n + 255) / 256, 256, 0, stream>>>(w2,    w2B,   n);
    }

    // 2) LN1(x) -> actB (bf16)
    ln_kernel<<<M_TOK, 256, 0, stream>>>(x, ln1_g, ln1_b, actB);

    // 3) qkv = LN1(x) @ w_qkv  -> qkvB (bf16)
    gemm_bf16_kernel<false,false,false,false,true>
        <<<dim3(C3_DIM / GBN, M_TOK / GBM), 256, 0, stream>>>(
        actB, wqkvB, C3_DIM, C_DIM, nullptr, nullptr, nullptr, qkvB);

    // 4) causal flash attention -> actB (bf16)
    attn_kernel<<<dim3(B_NUM * H_NUM, T_LEN / 64), 128, 0, stream>>>(qkvB, actB);

    // 5) x1 = x + attn @ w_out  (fp32)
    gemm_bf16_kernel<false,false,true,true,false>
        <<<dim3(C_DIM / GBN, M_TOK / GBM), 256, 0, stream>>>(
        actB, woutB, C_DIM, C_DIM, nullptr, x, x1, nullptr);

    // 6) LN2(x1) -> actB (bf16)
    ln_kernel<<<M_TOK, 256, 0, stream>>>(x1, ln2_g, ln2_b, actB);

    // 7) h1 = gelu(LN2 @ w1 + b1) -> h1B (bf16)
    gemm_bf16_kernel<true,true,false,false,true>
        <<<dim3(F_DIM / GBN, M_TOK / GBM), 256, 0, stream>>>(
        actB, w1B, F_DIM, C_DIM, b1, nullptr, nullptr, h1B);

    // 8) out = x1 + h1 @ w2 + b2  (fp32)
    gemm_bf16_kernel<true,false,true,true,false>
        <<<dim3(C_DIM / GBN, M_TOK / GBM), 256, 0, stream>>>(
        h1B, w2B, C_DIM, F_DIM, b2, x1, out, nullptr);
}
